// Drugcell_45887430591138
// MI455X (gfx1250) — compile-verified, matrix operand order checked by reading, not run on hardware
//
#include <hip/hip_runtime.h>
#include <hip/hip_bf16.h>

// ---------------------------------------------------------------------------
// MI455X (gfx1250) implementation. wave32, WMMA bf16 for the big skinny GEMMs.
// Memory-bound floor: fc1_w (2.1 GB fp32) + mlp weights (512 MB) @ 23.3 TB/s.
// All kernels deterministic (no float atomics): K-split partials + ordered
// reductions.
// ---------------------------------------------------------------------------

typedef __bf16 bf16_t;
typedef bf16_t v16bf __attribute__((ext_vector_type(16)));
typedef float  v8f   __attribute__((ext_vector_type(8)));

#define NODE 36

// ------------------------- conv(3->32,k5,p2)+relu+maxpool2 ------------------
__global__ void conv_pool_kernel(const float* __restrict__ img,   // [32,3,256,256]
                                 const float* __restrict__ cw,    // [32,3,5,5]
                                 const float* __restrict__ cb,
                                 float* __restrict__ out) {       // [32,32,128,128]
    int i = blockIdx.x * blockDim.x + threadIdx.x;
    if (i >= 32 * 32 * 128 * 128) return;
    int w = i & 127, h = (i >> 7) & 127, c = (i >> 14) & 31, n = i >> 19;
    const float* wp = cw + c * 75;
    float mx = -3.0e38f;
    #pragma unroll
    for (int dy = 0; dy < 2; ++dy)
    #pragma unroll
    for (int dx = 0; dx < 2; ++dx) {
        int oh = 2 * h + dy, ow = 2 * w + dx;
        float acc = cb[c];
        for (int ic = 0; ic < 3; ++ic) {
            const float* ip = img + ((n * 3 + ic) << 16);
            #pragma unroll
            for (int ky = 0; ky < 5; ++ky) {
                int iy = oh + ky - 2;
                if (iy < 0 || iy > 255) continue;
                #pragma unroll
                for (int kx = 0; kx < 5; ++kx) {
                    int ix = ow + kx - 2;
                    if (ix < 0 || ix > 255) continue;
                    acc += ip[(iy << 8) + ix] * wp[ic * 25 + ky * 5 + kx];
                }
            }
        }
        mx = fmaxf(mx, fmaxf(acc, 0.0f));
    }
    out[i] = mx;
}

// ------------------------- WMMA GEMM: part[kblk] = X[32,Kc] @ W[Kc,16] ------
// One wave per block. A rows 0..31 => two accumulators sharing B fragment.
// K-split over grid.y into per-chunk partial buffers (deterministic reduce).
__global__ void gemm_wmma(const float* __restrict__ X,   // [32, K] row-major
                          const float* __restrict__ W,   // [K, N] row-major
                          float* __restrict__ part,      // [nKChunks, 32, N]
                          int K, int N, int kChunk) {
    int n0   = blockIdx.x * 16;
    int k0   = blockIdx.y * kChunk;
    int k1   = min(k0 + kChunk, K);
    int lane = threadIdx.x;          // 0..31
    int half = lane >> 4;            // 0 or 1
    int idx  = lane & 15;            // M (for A) / N (for B) within tile
    v8f c0 = {}, c1 = {};
    for (int kb = k0; kb < k1; kb += 32) {
        v16bf a0, a1, b;
        #pragma unroll
        for (int j = 0; j < 16; ++j) {
            // ISA 7.12.2 16-bit A layout: lanes<16 hold K=0..7,16..23;
            // lanes>=16 hold K=8..15,24..31 (2 elems per VGPR).
            int kk = kb + ((j < 8) ? (8 * half + j) : (16 + 8 * half + (j - 8)));
            a0[j] = (bf16_t)X[idx * K + kk];
            a1[j] = (bf16_t)X[(idx + 16) * K + kk];
            b[j]  = (bf16_t)W[kk * N + n0 + idx];
        }
        if (kb + 64 < k1)   // keep the weight stream ahead in GL2
            __builtin_prefetch(&W[(kb + 64) * N + n0 + idx], 0, 1);
        c0 = __builtin_amdgcn_wmma_f32_16x16x32_bf16(false, a0, false, b,
                                                     (short)0, c0, false, false);
        c1 = __builtin_amdgcn_wmma_f32_16x16x32_bf16(false, a1, false, b,
                                                     (short)0, c1, false, false);
    }
    float* p = part + (size_t)blockIdx.y * 32 * N;
    #pragma unroll
    for (int r = 0; r < 8; ++r) {    // C/D: m = r + 8*half, n = idx
        int m = r + 8 * half;
        p[m * N + n0 + idx]        = c0[r];
        p[(16 + m) * N + n0 + idx] = c1[r];
    }
}

// ------------------------- ordered chunk-reduce + bias + relu ---------------
__global__ void reduce_bias_act(const float* __restrict__ part,
                                const float* __restrict__ bias,
                                float* __restrict__ out, int N, int nchunks,
                                int relu_flag) {
    int i = blockIdx.x * blockDim.x + threadIdx.x;
    if (i >= 32 * N) return;
    float s = 0.0f;
    for (int c = 0; c < nchunks; ++c) s += part[(size_t)c * 32 * N + i];
    s += bias[i % N];
    out[i] = relu_flag ? fmaxf(s, 0.0f) : s;
}

// ------------------------- [32,K] @ [K,36] + b (+optional softmax) ----------
__global__ void dense36(const float* __restrict__ X, const float* __restrict__ W,
                        const float* __restrict__ b, float* __restrict__ out,
                        int K, int do_softmax) {
    int row = blockIdx.x;        // 32 rows
    int col = threadIdx.x;       // blockDim 64, cols < 36 active
    __shared__ float sm[NODE];
    float acc = 0.0f;
    if (col < NODE) {
        acc = b[col];
        for (int k = 0; k < K; ++k) acc += X[row * K + k] * W[k * NODE + col];
        sm[col] = acc;
    }
    __syncthreads();
    if (col < NODE) {
        if (do_softmax) {
            float mx = -3.0e38f;
            for (int j = 0; j < NODE; ++j) mx = fmaxf(mx, sm[j]);
            float s = 0.0f;
            for (int j = 0; j < NODE; ++j) s += expf(sm[j] - mx);
            out[row * NODE + col] = expf(acc - mx) / s;
        } else {
            out[row * NODE + col] = acc;
        }
    }
}

// --------- semantic attention + CLIP loss (writes loss2, cell_cat) ----------
__global__ void att_clip_kernel(const float* __restrict__ rna1,
                                const float* __restrict__ cnv1,
                                const float* __restrict__ pic1,
                                const float* __restrict__ att_w1,  // [36,128]
                                const float* __restrict__ att_b1,  // [128]
                                const float* __restrict__ att_w2,  // [128,1]
                                const float* __restrict__ logit_scale,
                                float* __restrict__ cell_cat,      // [32,72]
                                float* __restrict__ loss2_out) {
    __shared__ float sv[64], beta[2];
    __shared__ float pn[32][NODE], cn[32][NODE];
    __shared__ float lg[32][32];
    __shared__ float red[64];
    int t = threadIdx.x;  // 256 threads
    if (t < 64) {
        int v = t >> 5, bb = t & 31;
        const float* zrow = (v == 0 ? rna1 : cnv1) + bb * NODE;
        float s = 0.0f;
        for (int j = 0; j < 128; ++j) {
            float h = att_b1[j];
            for (int i = 0; i < NODE; ++i) h += zrow[i] * att_w1[i * 128 + j];
            s += tanhf(h) * att_w2[j];
        }
        sv[t] = s;
    }
    __syncthreads();
    if (t == 0) {
        float w0 = 0.0f, w1 = 0.0f;
        for (int b = 0; b < 32; ++b) { w0 += sv[b]; w1 += sv[32 + b]; }
        w0 *= (1.0f / 32.0f); w1 *= (1.0f / 32.0f);
        float mx = fmaxf(w0, w1);
        float e0 = expf(w0 - mx), e1 = expf(w1 - mx), inv = 1.0f / (e0 + e1);
        beta[0] = e0 * inv; beta[1] = e1 * inv;
    }
    __syncthreads();
    if (t < 32) {
        float cc[NODE], pp[NODE], nc = 0.0f, np = 0.0f;
        for (int j = 0; j < NODE; ++j) {
            cc[j] = beta[0] * rna1[t * NODE + j] + beta[1] * cnv1[t * NODE + j];
            pp[j] = pic1[t * NODE + j];
            nc += cc[j] * cc[j]; np += pp[j] * pp[j];
        }
        float ic = 1.0f / sqrtf(nc), ip = 1.0f / sqrtf(np);
        for (int j = 0; j < NODE; ++j) {
            cn[t][j] = cc[j] * ic; pn[t][j] = pp[j] * ip;
            cell_cat[t * 72 + j]      = cn[t][j];
            cell_cat[t * 72 + 36 + j] = pn[t][j];
        }
    }
    __syncthreads();
    float scale = logit_scale[0];
    for (int p = t; p < 1024; p += 256) {
        int r = p >> 5, c = p & 31;
        float s = 0.0f;
        for (int j = 0; j < NODE; ++j) s += pn[r][j] * cn[c][j];
        lg[r][c] = scale * s;
    }
    __syncthreads();
    if (t < 64) {
        int r = t & 31;
        float mx = -3.0e38f, s = 0.0f;
        if (t < 32) {
            for (int c = 0; c < 32; ++c) mx = fmaxf(mx, lg[r][c]);
            for (int c = 0; c < 32; ++c) s += expf(lg[r][c] - mx);
        } else {
            for (int rr = 0; rr < 32; ++rr) mx = fmaxf(mx, lg[rr][r]);
            for (int rr = 0; rr < 32; ++rr) s += expf(lg[rr][r] - mx);
        }
        red[t] = (logf(s) + mx) - lg[r][r];
    }
    __syncthreads();
    if (t == 0) {
        float a = 0.0f, b = 0.0f;
        for (int i = 0; i < 32; ++i) { a += red[i]; b += red[32 + i]; }
        loss2_out[0] = 0.5f * (a / 32.0f + b / 32.0f);
    }
}

// ------------------------- drug GCN (one block per drug) --------------------
// out = Hn @ (Hn @ ((X W) W2)) using associativity; relu, max over atoms.
__global__ void gcn_kernel(const float* __restrict__ drug_fea,  // [128,64,75]
                           const float* __restrict__ drug_adj,  // [128,64,64]
                           const float* __restrict__ gcn_w,     // [75,288]
                           const float* __restrict__ gcn_w2,    // [288,72]
                           float* __restrict__ dfea) {          // [128,72]
    int d = blockIdx.x, t = threadIdx.x;   // 256 threads
    __shared__ float Hn[64 * 64];
    __shared__ float dinv[64];
    __shared__ float Yb[64 * 32];
    __shared__ float P[64 * 72];
    __shared__ float Z[64 * 72];
    const float* A = drug_adj + d * 4096;
    const float* X = drug_fea + d * 64 * 75;
    for (int i = t; i < 4096; i += 256) {
        int r = i >> 6, c = i & 63;
        Hn[i] = (r == c) ? 1.0f : A[c * 64 + r];   // A^T with forced diag 1
    }
    __syncthreads();
    if (t < 64) {
        float s = 0.0f;
        for (int c = 0; c < 64; ++c) s += Hn[t * 64 + c];
        dinv[t] = (s == 0.0f) ? 0.0f : 1.0f / s;
    }
    __syncthreads();
    for (int i = t; i < 4096; i += 256) Hn[i] *= dinv[i >> 6];
    for (int i = t; i < 64 * 72; i += 256) P[i] = 0.0f;
    __syncthreads();
    for (int blk = 0; blk < 9; ++blk) {      // 288 = 9 * 32 cols of X@gcn_w
        for (int i = t; i < 64 * 32; i += 256) {
            int a = i >> 5, j = i & 31;
            float s = 0.0f;
            for (int f = 0; f < 75; ++f)
                s += X[a * 75 + f] * gcn_w[f * 288 + blk * 32 + j];
            Yb[i] = s;
        }
        __syncthreads();
        for (int i = t; i < 64 * 72; i += 256) {
            int a = i / 72, o = i % 72;
            float s = P[i];
            for (int j = 0; j < 32; ++j)
                s += Yb[a * 32 + j] * gcn_w2[(blk * 32 + j) * 72 + o];
            P[i] = s;
        }
        __syncthreads();
    }
    for (int i = t; i < 64 * 72; i += 256) {       // Z = Hn @ P
        int a = i / 72, o = i % 72;
        float s = 0.0f;
        for (int m = 0; m < 64; ++m) s += Hn[a * 64 + m] * P[m * 72 + o];
        Z[i] = s;
    }
    __syncthreads();
    for (int i = t; i < 64 * 72; i += 256) {       // relu(Hn @ Z) -> P
        int a = i / 72, o = i % 72;
        float s = 0.0f;
        for (int m = 0; m < 64; ++m) s += Hn[a * 64 + m] * Z[m * 72 + o];
        P[i] = fmaxf(s, 0.0f);
    }
    __syncthreads();
    if (t < 72) {
        float mx = -3.0e38f;
        for (int a = 0; a < 64; ++a) mx = fmaxf(mx, P[a * 72 + t]);
        dfea[d * 72 + t] = mx;
    }
}

// ------------------------- pair head + per-pair CE terms --------------------
__global__ void pair_head(const float* __restrict__ cell_cat,
                          const float* __restrict__ dfea,
                          const int* __restrict__ index_list,
                          const int* __restrict__ target,
                          const float* __restrict__ w1, const float* __restrict__ b1,
                          const float* __restrict__ w2, const float* __restrict__ b2,
                          float* __restrict__ d_out, float* __restrict__ pl) {
    int p = blockIdx.x * blockDim.x + threadIdx.x;
    if (p >= 4096) return;
    const float* cc = cell_cat + index_list[2 * p] * 72;
    const float* df = dfea + index_list[2 * p + 1] * 72;
    float h[NODE];
    #pragma unroll 4
    for (int o = 0; o < NODE; ++o) {
        float s = b1[o];
        for (int j = 0; j < 72; ++j) s += cc[j] * w1[j * NODE + o];
        for (int j = 0; j < 72; ++j) s += df[j] * w1[(72 + j) * NODE + o];
        h[o] = fmaxf(s, 0.0f);
    }
    float l0 = b2[0], l1 = b2[1];
    for (int o = 0; o < NODE; ++o) { l0 += h[o] * w2[o * 2]; l1 += h[o] * w2[o * 2 + 1]; }
    float mx = fmaxf(l0, l1);
    float e0 = expf(l0 - mx), e1 = expf(l1 - mx), inv = 1.0f / (e0 + e1);
    float p0 = e0 * inv, p1 = e1 * inv;
    d_out[2 + 2 * p] = p0;
    d_out[3 + 2 * p] = p1;
    int tg = target[p];
    d_out[2 + 8192 + p] = (float)tg;
    // reference applies log_softmax again to the probabilities
    float m2 = fmaxf(p0, p1);
    float lse = m2 + logf(expf(p0 - m2) + expf(p1 - m2));
    pl[p] = lse - (tg == 0 ? p0 : p1);
}

__global__ void loss_reduce(const float* __restrict__ pl, float* __restrict__ out0) {
    __shared__ float red[256];
    int t = threadIdx.x;
    float s = 0.0f;
    for (int i = t; i < 4096; i += 256) s += pl[i];
    red[t] = s;
    __syncthreads();
    for (int w = 128; w > 0; w >>= 1) {
        if (t < w) red[t] += red[t + w];
        __syncthreads();
    }
    if (t == 0) out0[0] = red[0] / 4096.0f;
}

// ---------------------------------------------------------------------------
extern "C" void kernel_launch(void* const* d_in, const int* in_sizes, int n_in,
                              void* d_out, int out_size, void* d_ws, size_t ws_size,
                              hipStream_t stream) {
    const float* pic_dim     = (const float*)d_in[0];
    const float* rna_seq     = (const float*)d_in[1];
    const float* gene_cnv    = (const float*)d_in[2];
    const float* drug_fea    = (const float*)d_in[3];
    const float* drug_adj    = (const float*)d_in[4];
    const int*   index_list  = (const int*)  d_in[5];
    const int*   target      = (const int*)  d_in[6];
    const float* logit_scale = (const float*)d_in[7];
    const float* conv_w  = (const float*)d_in[8];
    const float* conv_b  = (const float*)d_in[9];
    const float* fc1_w   = (const float*)d_in[10];
    const float* fc1_b   = (const float*)d_in[11];
    const float* fc2_w   = (const float*)d_in[12];
    const float* fc2_b   = (const float*)d_in[13];
    const float* mlp1_w1 = (const float*)d_in[14];
    const float* mlp1_b1 = (const float*)d_in[15];
    const float* mlp1_w2 = (const float*)d_in[16];
    const float* mlp1_b2 = (const float*)d_in[17];
    const float* mlp2_w1 = (const float*)d_in[18];
    const float* mlp2_b1 = (const float*)d_in[19];
    const float* mlp2_w2 = (const float*)d_in[20];
    const float* mlp2_b2 = (const float*)d_in[21];
    const float* att_w1  = (const float*)d_in[22];
    const float* att_b1  = (const float*)d_in[23];
    const float* att_w2  = (const float*)d_in[24];
    const float* end_w1  = (const float*)d_in[25];
    const float* end_b1  = (const float*)d_in[26];
    const float* end_w2  = (const float*)d_in[27];
    const float* end_b2  = (const float*)d_in[28];
    const float* gcn_w   = (const float*)d_in[29];
    const float* gcn_w2  = (const float*)d_in[30];

    float* ws  = (float*)d_ws;
    float* out = (float*)d_out;

    // workspace layout (floats)
    size_t o_pool = 0;                          // 32*524288
    size_t o_fc1p = o_pool + 16777216;          // 64 chunks * 32*1024
    size_t o_rnap = o_fc1p + 2097152;           // 4 chunks * 32*4000
    size_t o_cnvp = o_rnap + 512000;
    size_t o_fc1a = o_cnvp + 512000;            // 32*1024
    size_t o_h1r  = o_fc1a + 32768;             // 32*4000
    size_t o_h1c  = o_h1r + 128000;
    size_t o_rna1 = o_h1c + 128000;             // 32*36 each
    size_t o_cnv1 = o_rna1 + 1152;
    size_t o_pic1 = o_cnv1 + 1152;
    size_t o_ccat = o_pic1 + 1152;              // 32*72
    size_t o_dfea = o_ccat + 2304;              // 128*72
    size_t o_plos = o_dfea + 9216;              // 4096

    float* pool = ws + o_pool;  float* fc1p = ws + o_fc1p;
    float* rnap = ws + o_rnap;  float* cnvp = ws + o_cnvp;
    float* fc1a = ws + o_fc1a;  float* h1r  = ws + o_h1r;
    float* h1c  = ws + o_h1c;   float* rna1 = ws + o_rna1;
    float* cnv1 = ws + o_cnv1;  float* pic1 = ws + o_pic1;
    float* ccat = ws + o_ccat;  float* dfea = ws + o_dfea;
    float* plos = ws + o_plos;

    // image branch: conv+relu+pool -> [32, 524288]
    conv_pool_kernel<<<65536, 256, 0, stream>>>(pic_dim, conv_w, conv_b, pool);

    // omics L1 GEMMs: [32,16000]@[16000,4000], K-split 4 -> 1000 waves each
    gemm_wmma<<<dim3(250, 4), 32, 0, stream>>>(rna_seq,  mlp1_w1, rnap, 16000, 4000, 4000);
    gemm_wmma<<<dim3(250, 4), 32, 0, stream>>>(gene_cnv, mlp2_w1, cnvp, 16000, 4000, 4000);
    // fc1 GEMM: [32,524288]@[524288,1024], K-split 64 -> 4096 waves (BW-bound)
    gemm_wmma<<<dim3(64, 64), 32, 0, stream>>>(pool, fc1_w, fc1p, 524288, 1024, 8192);

    reduce_bias_act<<<500, 256, 0, stream>>>(rnap, mlp1_b1, h1r, 4000, 4, 1);
    reduce_bias_act<<<500, 256, 0, stream>>>(cnvp, mlp2_b1, h1c, 4000, 4, 1);
    reduce_bias_act<<<128, 256, 0, stream>>>(fc1p, fc1_b, fc1a, 1024, 64, 1);

    dense36<<<32, 64, 0, stream>>>(h1r,  mlp1_w2, mlp1_b2, rna1, 4000, 1);
    dense36<<<32, 64, 0, stream>>>(h1c,  mlp2_w2, mlp2_b2, cnv1, 4000, 1);
    dense36<<<32, 64, 0, stream>>>(fc1a, fc2_w,   fc2_b,   pic1, 1024, 0);

    att_clip_kernel<<<1, 256, 0, stream>>>(rna1, cnv1, pic1, att_w1, att_b1,
                                           att_w2, logit_scale, ccat, out + 1);

    gcn_kernel<<<128, 256, 0, stream>>>(drug_fea, drug_adj, gcn_w, gcn_w2, dfea);

    pair_head<<<32, 128, 0, stream>>>(ccat, dfea, index_list, target,
                                      end_w1, end_b1, end_w2, end_b2, out, plos);
    loss_reduce<<<1, 256, 0, stream>>>(plos, out);
}
